// LSTMSimple_84782654423270
// MI455X (gfx1250) — matrix operand order repaired and checked by reading, not compiled, not run on hardware
//
#include <hip/hip_runtime.h>
#include <math.h>

// ---------------------------------------------------------------------------
// Fused 2-layer LSTM (B=128, T=1024, F=128, H=256) + dense head for MI455X.
// 8 workgroups x 16 batch rows; each runs the full sequential scan privately.
// bf16 WMMA (v_wmma_f32_16x16x32_bf16), fp32 accumulators, fp32 cell state.
// Weights pre-transposed to [N,K] bf16 so B-fragments are K-contiguous 32B
// global loads; A-fragments come from LDS (2x ds_load_b128).
// Round-4: software-pipeline each GEMM segment one K-chunk ahead so weight
// loads (L2) overlap WMMAs (partial s_wait_loadcnt instead of full drains).
// Keeps: opaque-zero LICM blocker (round 3), per-chunk sched fence bounding
// in-flight fragments to two chunks (no spills).
// ---------------------------------------------------------------------------

typedef __attribute__((ext_vector_type(16))) __bf16 bf16x16;
typedef __attribute__((ext_vector_type(8)))  __bf16 bf16x8;
typedef __attribute__((ext_vector_type(8)))  float  f32x8;

#define B_  128
#define T_  1024
#define F_  128
#define H_  256
#define G_  1024   // 4*H

#if defined(__has_builtin)
#if __has_builtin(__builtin_amdgcn_sched_barrier)
#define SCHED_FENCE() __builtin_amdgcn_sched_barrier(0)
#endif
#endif
#ifndef SCHED_FENCE
#define SCHED_FENCE()
#endif

__device__ __forceinline__ float sigf(float v) { return 1.0f / (1.0f + __expf(-v)); }

// Returns 0 at runtime, but the optimizer cannot prove it (opaque asm that
// depends on t). Added to weight base pointers each step: blocks LICM from
// hoisting the t-invariant weight loads out of the scan loop (spill cause).
__device__ __forceinline__ int opaque_zero(int t) {
  int tv;
  asm("s_and_b32 %0, %1, 0" : "=s"(tv) : "s"(t));
  return tv;
}

// A-fragment (16x32 bf16, M x K) from row-major [16, K] bf16 buffer (LDS).
// Lanes 0-15 hold K 0..15; lanes 16-31 hold K 16..31 (matches B-fragment);
// row M = lane & 15.
__device__ __forceinline__ bf16x16 load_a_frag(const __bf16* base, int K, int kc,
                                               int m, int hi) {
  const __bf16* p = base + m * K + kc * 32 + hi * 16;
  bf16x8 lo = *(const bf16x8*)(p);
  bf16x8 up = *(const bf16x8*)(p + 8);
  return __builtin_shufflevector(lo, up, 0, 1, 2, 3, 4, 5, 6, 7,
                                         8, 9, 10, 11, 12, 13, 14, 15);
}

// B-fragment (32x16 bf16, K x N) from pre-transposed weights Wt[N][K] bf16.
// Lane n = col; lanes 0-15 hold K 0..15, lanes 16-31 hold K 16..31 of chunk:
// 16 K-contiguous bf16 = one 32-byte load (2x global_load_b128).
__device__ __forceinline__ bf16x16 load_b_frag(const __bf16* Wt, int K, int n,
                                               int kc, int hi) {
  return *(const bf16x16*)(Wt + (size_t)n * K + kc * 32 + hi * 16);
}

// One GEMM segment: acc[g] += A(16xK) x Wt-cols[col + g*256], K = NK*32.
// Software-pipelined one K-chunk ahead: chunk kc+1's loads are issued before
// chunk kc's WMMAs; the fence after the WMMAs caps in-flight data at 2 chunks.
template <int NK>
__device__ __forceinline__ void gemm_seg(const __bf16* Alds, int KA,
                                         const __bf16* Wt, int KW,
                                         int col, int hi, int nn, f32x8* acc) {
  bf16x16 a_cur = load_a_frag(Alds, KA, 0, nn, hi);
  bf16x16 b_cur[4];
#pragma unroll
  for (int g = 0; g < 4; ++g) b_cur[g] = load_b_frag(Wt, KW, col + g * 256, 0, hi);

#pragma unroll
  for (int kc = 0; kc < NK; ++kc) {
    bf16x16 a_nxt;
    bf16x16 b_nxt[4];
    if (kc + 1 < NK) {                       // issue next chunk's loads first
      a_nxt = load_a_frag(Alds, KA, kc + 1, nn, hi);
#pragma unroll
      for (int g = 0; g < 4; ++g)
        b_nxt[g] = load_b_frag(Wt, KW, col + g * 256, kc + 1, hi);
    }
#pragma unroll
    for (int g = 0; g < 4; ++g)
      acc[g] = __builtin_amdgcn_wmma_f32_16x16x32_bf16(
          false, a_cur, false, b_cur[g], (short)0, acc[g], false, false);
    SCHED_FENCE();                           // kc+2 loads cannot hoist above
    if (kc + 1 < NK) {
      a_cur = a_nxt;
#pragma unroll
      for (int g = 0; g < 4; ++g) b_cur[g] = b_nxt[g];
    }
  }
}

// out[n*K + k] = bf16(in[k*N + n])   (fp32 [K,N] -> bf16 [N,K])
__global__ void transpose_to_bf16(const float* __restrict__ in,
                                  __bf16* __restrict__ out, int N, int K) {
  int i = blockIdx.x * blockDim.x + threadIdx.x;
  if (i >= N * K) return;
  int n = i / K;
  int k = i - n * K;
  out[i] = (__bf16)in[(size_t)k * N + n];
}

__global__ void __launch_bounds__(512, 1)
lstm_fused(const float* __restrict__ x,
           const __bf16* __restrict__ W1t, const __bf16* __restrict__ U1t,
           const __bf16* __restrict__ W2t, const __bf16* __restrict__ U2t,
           const __bf16* __restrict__ Wd1t,
           const float* __restrict__ b1,  const float* __restrict__ b2,
           const float* __restrict__ bd1, const float* __restrict__ Wd2,
           const float* __restrict__ bd2,
           float* __restrict__ out) {
  __shared__ __bf16 sx[16 * F_];    // staged x tile, bf16          (4 KB)
  __shared__ __bf16 sh1[16 * H_];   // layer-1 hidden state, bf16   (8 KB)
  __shared__ __bf16 sh2[16 * H_];   // layer-2 hidden state, bf16   (8 KB)
  __shared__ float  sd1[16 * 128];  // dense-1 activations, fp32    (8 KB)

  const int tid  = threadIdx.x;
  const int wv   = tid >> 5;    // wave 0..15
  const int lane = tid & 31;
  const int hi   = lane >> 4;   // half-wave select
  const int nn   = lane & 15;   // column within tile / A row
  const int gb0  = blockIdx.x * 16;
  const int col  = wv * 16 + nn;

  for (int i = tid; i < 16 * H_; i += 512) {
    sh1[i] = (__bf16)0.0f;
    sh2[i] = (__bf16)0.0f;
  }

  // Per-wave gate biases: wave wv owns gate columns wv*16 + g*256 (+nn).
  float bias1[4], bias2[4];
#pragma unroll
  for (int g = 0; g < 4; ++g) {
    bias1[g] = b1[col + g * 256];
    bias2[g] = b2[col + g * 256];
  }

  f32x8 c1 = {};  // cell state tile (16 rows x cols wv*16..wv*16+15)
  f32x8 c2 = {};

  __syncthreads();

  for (int t = 0; t < T_; ++t) {
    // Opaque zero: weight pointers become formally t-variant (blocks LICM).
    const int tv = opaque_zero(t);
    const __bf16* W1p = W1t + tv;
    const __bf16* U1p = U1t + tv;
    const __bf16* W2p = W2t + tv;
    const __bf16* U2p = U2t + tv;

    // ---- stage x[:, t, :] tile (16 x 128 fp32 -> bf16 in LDS) -------------
    {
      int row = tid >> 5;   // 0..15
      const float4 v = *(const float4*)(x + (size_t)(gb0 + row) * (T_ * F_) +
                                        (size_t)t * F_ + lane * 4);
      sx[row * F_ + lane * 4 + 0] = (__bf16)v.x;
      sx[row * F_ + lane * 4 + 1] = (__bf16)v.y;
      sx[row * F_ + lane * 4 + 2] = (__bf16)v.z;
      sx[row * F_ + lane * 4 + 3] = (__bf16)v.w;
    }
    __syncthreads();

    // ---- layer 1: gates = x_t @ W1 + h1 @ U1 + b1 -------------------------
    f32x8 acc[4];
#pragma unroll
    for (int g = 0; g < 4; ++g) {
      f32x8 a = {};
#pragma unroll
      for (int r = 0; r < 8; ++r) a[r] = bias1[g];
      acc[g] = a;
    }
    gemm_seg<F_ / 32>(sx,  F_, W1p, F_, col, hi, nn, acc);   // x  @ W1
    gemm_seg<H_ / 32>(sh1, H_, U1p, H_, col, hi, nn, acc);   // h1 @ U1
    __syncthreads();   // all waves done reading old sh1

    // ---- layer-1 pointwise update (registers), publish new h1 -------------
#pragma unroll
    for (int r = 0; r < 8; ++r) {
      float iv = sigf(acc[0][r]);
      float fv = sigf(acc[1][r]);
      float gv = tanhf(acc[2][r]);
      float ov = sigf(acc[3][r]);
      float cn = fv * c1[r] + iv * gv;
      c1[r] = cn;
      sh1[(r + hi * 8) * H_ + col] = (__bf16)(ov * tanhf(cn));
    }
    __syncthreads();

    // ---- layer 2: gates = h1 @ W2 + h2 @ U2 + b2 --------------------------
#pragma unroll
    for (int g = 0; g < 4; ++g) {
      f32x8 a = {};
#pragma unroll
      for (int r = 0; r < 8; ++r) a[r] = bias2[g];
      acc[g] = a;
    }
    gemm_seg<H_ / 32>(sh1, H_, W2p, H_, col, hi, nn, acc);   // h1 @ W2
    gemm_seg<H_ / 32>(sh2, H_, U2p, H_, col, hi, nn, acc);   // h2 @ U2
    __syncthreads();   // all waves done reading old sh2

#pragma unroll
    for (int r = 0; r < 8; ++r) {
      float iv = sigf(acc[0][r]);
      float fv = sigf(acc[1][r]);
      float gv = tanhf(acc[2][r]);
      float ov = sigf(acc[3][r]);
      float cn = fv * c2[r] + iv * gv;
      c2[r] = cn;
      sh2[(r + hi * 8) * H_ + col] = (__bf16)(ov * tanhf(cn));
    }
    __syncthreads();
  }

  // ---- dense head: relu(h2 @ Wd1 + bd1) -> relu(@ Wd2 + bd2) --------------
  if (wv < 8) {                       // 8 waves cover the 128 output columns
    f32x8 accd = {};
#pragma unroll
    for (int r = 0; r < 8; ++r) accd[r] = bd1[col];
#pragma unroll
    for (int kc = 0; kc < H_ / 32; ++kc) {
      bf16x16 a = load_a_frag(sh2, H_, kc, nn, hi);
      bf16x16 b = load_b_frag(Wd1t, H_, col, kc, hi);
      accd = __builtin_amdgcn_wmma_f32_16x16x32_bf16(
          false, a, false, b, (short)0, accd, false, false);
      SCHED_FENCE();
    }
#pragma unroll
    for (int r = 0; r < 8; ++r)
      sd1[(r + hi * 8) * 128 + col] = fmaxf(accd[r], 0.0f);
  }
  __syncthreads();

  if (tid < 16) {
    float s = bd2[0];
#pragma unroll 4
    for (int k = 0; k < 128; ++k) s += sd1[tid * 128 + k] * Wd2[k];
    out[gb0 + tid] = fmaxf(s, 0.0f);
  }
}

extern "C" void kernel_launch(void* const* d_in, const int* in_sizes, int n_in,
                              void* d_out, int out_size, void* d_ws, size_t ws_size,
                              hipStream_t stream) {
  (void)in_sizes; (void)n_in; (void)out_size; (void)ws_size;
  const float* x   = (const float*)d_in[0];
  const float* W1  = (const float*)d_in[1];
  const float* U1  = (const float*)d_in[2];
  const float* b1  = (const float*)d_in[3];
  const float* W2  = (const float*)d_in[4];
  const float* U2  = (const float*)d_in[5];
  const float* b2  = (const float*)d_in[6];
  const float* Wd1 = (const float*)d_in[7];
  const float* bd1 = (const float*)d_in[8];
  const float* Wd2 = (const float*)d_in[9];
  const float* bd2 = (const float*)d_in[10];
  float* out = (float*)d_out;

  // Workspace: pre-transposed bf16 weights ([N,K]), ~1.9 MB total.
  __bf16* W1t  = (__bf16*)d_ws;                 // [1024][128]
  __bf16* U1t  = W1t + 1024 * 128;              // [1024][256]
  __bf16* W2t  = U1t + 1024 * 256;              // [1024][256]
  __bf16* U2t  = W2t + 1024 * 256;              // [1024][256]
  __bf16* Wd1t = U2t + 1024 * 256;              // [128][256]

  auto launchT = [&](const float* in, __bf16* o, int N, int K) {
    int total = N * K;
    transpose_to_bf16<<<(total + 255) / 256, 256, 0, stream>>>(in, o, N, K);
  };
  launchT(W1,  W1t,  1024, 128);
  launchT(U1,  U1t,  1024, 256);
  launchT(W2,  W2t,  1024, 256);
  launchT(U2,  U2t,  1024, 256);
  launchT(Wd1, Wd1t, 128,  256);

  lstm_fused<<<8, 512, 0, stream>>>(x, W1t, U1t, W2t, U2t, Wd1t,
                                    b1, b2, bd1, Wd2, bd2, out);
}
